// my_MultiheadAttention_15083925144173
// MI455X (gfx1250) — matrix operand order repaired and checked by reading, not compiled, not run on hardware
//
#include <hip/hip_runtime.h>

#ifndef __has_builtin
#define __has_builtin(x) 0
#endif

#define LSEQ   4096
#define DMODEL 1024
#define NHEAD  16
#define DK     64

typedef __attribute__((ext_vector_type(16))) _Float16     v16h;
typedef __attribute__((ext_vector_type(8)))  _Float16     v8h;
typedef __attribute__((ext_vector_type(8)))  float        v8f;
typedef __attribute__((ext_vector_type(4)))  unsigned int u32x4;
typedef __attribute__((ext_vector_type(4)))  int          i32x4;
typedef __attribute__((ext_vector_type(8)))  int          i32x8;

// --- CDNA5 feature selection (device pass only; host pass sees fallbacks) ---
#if __has_builtin(__builtin_amdgcn_tensor_load_to_lds) && __has_builtin(__builtin_amdgcn_s_wait_tensorcnt)
#define USE_TDM 1
#elif __has_builtin(__builtin_amdgcn_global_load_async_to_lds_b128)
#define USE_ASYNC 1
#endif
#if __has_builtin(__builtin_amdgcn_update_dpp)
#define HAVE_DPP 1
#endif

// ---- WMMA helpers (CDNA5 wave32, V_WMMA_F32_16X16X32_F16) ----

__device__ __forceinline__ v8f wmma_f16(v16h a, v16h b, v8f c) {
  return __builtin_amdgcn_wmma_f32_16x16x32_f16(false, a, false, b, (short)0, c,
                                                false, false);
}

// A-matrix 16x32 f16 fragment: lane m = lane&15, kb = 16*(lane>>4),
// element i holds K = kb + i.  Caller passes p = base + m*lda + kb.
__device__ __forceinline__ v16h load_a_frag(const _Float16* p) {
  return *(const v16h*)(p);
}

// 16-lane (half-wave) max reduction: DPP ROW_XMASK on the VALU pipe, no LDS.
__device__ __forceinline__ float rowmax16(float v) {
#if defined(HAVE_DPP)
  int t;
  t = __builtin_amdgcn_update_dpp(0, __builtin_bit_cast(int, v), 0x161, 0xf, 0xf, true);
  v = fmaxf(v, __builtin_bit_cast(float, t));
  t = __builtin_amdgcn_update_dpp(0, __builtin_bit_cast(int, v), 0x162, 0xf, 0xf, true);
  v = fmaxf(v, __builtin_bit_cast(float, t));
  t = __builtin_amdgcn_update_dpp(0, __builtin_bit_cast(int, v), 0x164, 0xf, 0xf, true);
  v = fmaxf(v, __builtin_bit_cast(float, t));
  t = __builtin_amdgcn_update_dpp(0, __builtin_bit_cast(int, v), 0x168, 0xf, 0xf, true);
  v = fmaxf(v, __builtin_bit_cast(float, t));
#else
  v = fmaxf(v, __shfl_xor(v, 1));
  v = fmaxf(v, __shfl_xor(v, 2));
  v = fmaxf(v, __shfl_xor(v, 4));
  v = fmaxf(v, __shfl_xor(v, 8));
#endif
  return v;
}

// ---- Tensor Data Mover: build D# per ISA ch.8 and issue TENSOR_LOAD_TO_LDS ----
#if defined(USE_TDM)
__device__ __forceinline__ void tdm_load_tile(void* lds, const void* gsrc, i32x8 g1) {
  unsigned lofs = (unsigned)(unsigned long long)(__attribute__((address_space(3))) void*)lds;
  unsigned long long ga = (unsigned long long)gsrc;
  u32x4 g0;
  g0[0] = 1u;                                                   // count=1, user mode
  g0[1] = lofs;                                                 // lds_addr (bytes)
  g0[2] = (unsigned)ga;                                         // global_addr[31:0]
  g0[3] = (unsigned)((ga >> 32) & 0x01ffffffull) | 0x80000000u; // addr[56:32] | type=2
  i32x4 z4 = {0, 0, 0, 0};
#if defined(__clang_major__) && (__clang_major__ >= 23)
  i32x8 z8 = {0, 0, 0, 0, 0, 0, 0, 0};
  __builtin_amdgcn_tensor_load_to_lds(g0, g1, z4, z4, z8, 0);
#else
  __builtin_amdgcn_tensor_load_to_lds(g0, g1, z4, z4, 0);
#endif
}
#endif

// Stage one 64-key tile: K (contiguous 8KB) and V^T (64 rows x 64 keys, row stride LSEQ)
__device__ __forceinline__ void stage_issue(const _Float16* kSrc, const _Float16* vSrc,
                                            _Float16* kDst, _Float16* vDst,
                                            int tid, int wave) {
#if defined(USE_TDM)
  if (wave == 0) {
    // group1: data_size=2B; K: 1-D tile of 4096 elems; V: 64x64 tile, dim0 stride 4096
    const i32x8 G1K = {0x00010000, 0x10000000, 0x00010000, 0x10000000, 0, 4096, 0, 0};
    const i32x8 G1V = {0x00010000, 0x10000000, 0x00400000, 0x00400000, 64, 4096, 0, 0};
    tdm_load_tile(kDst, kSrc, G1K);
    tdm_load_tile(vDst, vSrc, G1V);
  }
  (void)tid;
#elif defined(USE_ASYNC)
#pragma unroll
  for (int i = 0; i < 4; ++i) {
    const int idx = tid + i * 128;
    __builtin_amdgcn_global_load_async_to_lds_b128(
        (const __attribute__((address_space(1))) void*)(kSrc + idx * 8),
        (__attribute__((address_space(3))) void*)(kDst + idx * 8), 0, 0);
    const int dc = idx >> 3, off = idx & 7;
    __builtin_amdgcn_global_load_async_to_lds_b128(
        (const __attribute__((address_space(1))) void*)(vSrc + (size_t)dc * LSEQ + off * 8),
        (__attribute__((address_space(3))) void*)(vDst + idx * 8), 0, 0);
  }
  (void)wave;
#else
#pragma unroll
  for (int i = 0; i < 4; ++i) {
    const int idx = tid + i * 128;
    ((uint4*)kDst)[idx] = ((const uint4*)kSrc)[idx];
    const int dc = idx >> 3, off = idx & 7;
    ((uint4*)vDst)[idx] = ((const uint4*)(vSrc + (size_t)dc * LSEQ))[off];
  }
  (void)wave;
#endif
}

__device__ __forceinline__ void stage_sync(int wave) {
#if defined(USE_TDM)
  if (wave == 0) __builtin_amdgcn_s_wait_tensorcnt(0);
#elif defined(USE_ASYNC)
  asm volatile("s_wait_asynccnt 0x0" ::: "memory");
#else
  (void)wave;
#endif
  __syncthreads();
}

// ---- elementwise f32 -> f16 conversion ----

__global__ void __launch_bounds__(256) cvt_f32_f16(const float* __restrict__ src,
                                                   _Float16* __restrict__ dst, int n) {
  int i = blockIdx.x * 256 + threadIdx.x;
  if (i < n) dst[i] = (_Float16)src[i];
}

// ---- GEMM: C[M][N] = A[M][K] * W[N][K]^T + bias[N], one 16x16 tile per wave ----
// OUT_MODE 0: f16 head-major [H][L][DK]  (Q, K)
// OUT_MODE 1: f16 head-transposed [H][DK][L]  (V, feeds TDM-friendly attention tiles)
// OUT_MODE 2: f32 [M][N]  (final projection)

template <int OUT_MODE>
__global__ void __launch_bounds__(256) gemm16_wmma(const _Float16* __restrict__ A,
                                                   const _Float16* __restrict__ W,
                                                   const float* __restrict__ bias,
                                                   void* __restrict__ out) {
  const int K = DMODEL, N = DMODEL;
  const int lane = threadIdx.x & 31;
  const int wave = threadIdx.x >> 5;
  const int tile = blockIdx.x * 8 + wave;
  const int tn = tile & (N / 16 - 1);
  const int tm = tile / (N / 16);
  const int m0 = tm << 4, n0 = tn << 4;
  const int ml = lane & 15, hl = lane >> 4;

  const _Float16* Ap = A + (size_t)(m0 + ml) * K + (hl << 4);
  const _Float16* Wp = W + (size_t)(n0 + ml) * K + (hl << 4);

  v8f acc = {};
#pragma unroll 4
  for (int k = 0; k < K; k += 32) {
    __builtin_prefetch(Ap + k + 512, 0, 1);   // global_prefetch_b8, counter-free
    __builtin_prefetch(Wp + k + 512, 0, 1);
    v16h a = load_a_frag(Ap + k);
    v16h b = *(const v16h*)(Wp + k);
    acc = wmma_f16(a, b, acc);
  }

  const float bv = bias[n0 + ml];
#pragma unroll
  for (int r = 0; r < 8; ++r) {
    const int row = m0 + r + (hl << 3);   // C layout: VGPR r -> M = r + 8*(lane>=16)
    const int col = n0 + ml;
    const float v = acc[r] + bv;
    if (OUT_MODE == 2) {
      ((float*)out)[(size_t)row * N + col] = v;
    } else if (OUT_MODE == 1) {
      ((_Float16*)out)[((size_t)(col >> 6) * DK + (col & 63)) * LSEQ + row] = (_Float16)v;
    } else {
      ((_Float16*)out)[((size_t)(col >> 6) * LSEQ + row) * DK + (col & 63)] = (_Float16)v;
    }
  }
}

// ---- fused flash attention: one head x 64 query rows per block (4 waves) ----
// TDM/async double-buffered K,V tiles; denominator accumulated via ones-column WMMA.

__global__ void __launch_bounds__(128) attn_wmma(const _Float16* __restrict__ Qh,
                                                 const _Float16* __restrict__ Kh,
                                                 const _Float16* __restrict__ VhT,
                                                 _Float16* __restrict__ Oh) {
  __shared__ __align__(32) _Float16 Ks[2][64 * 64];   // [key][d]
  __shared__ __align__(32) _Float16 Vs[2][64 * 64];   // [d][key] (already transposed)
  __shared__ __align__(32) _Float16 Ps[4 * 16 * 64];  // per-wave P staging [16][64]

  const int tid = threadIdx.x;
  const int lane = tid & 31, wave = tid >> 5;
  const int h = blockIdx.y;
  const int q0 = blockIdx.x * 64 + wave * 16;
  const int ml = lane & 15, hl = lane >> 4;

  // Q fragments, softmax scale 1/sqrt(64)=0.125 folded in (exact in f16)
  const _Float16* Qp = Qh + ((size_t)h * LSEQ + q0 + ml) * DK + (hl << 4);
  v16h aq0 = load_a_frag(Qp);
  v16h aq1 = load_a_frag(Qp + 32);
#pragma unroll
  for (int i = 0; i < 16; ++i) {
    aq0[i] *= (_Float16)0.125f;
    aq1[i] *= (_Float16)0.125f;
  }

  // ones-column B fragment: accumulates the softmax denominator on the matrix pipe
  v16h bone;
#pragma unroll
  for (int i = 0; i < 16; ++i) bone[i] = (ml == 0) ? (_Float16)1.0f : (_Float16)0.0f;

  v8f o[5];          // o[0..3]: unnormalized O tiles; o[4]: row-sum tile (col 0)
  float m_r[8];
  {
    v8f z = {};
#pragma unroll
    for (int d = 0; d < 5; ++d) o[d] = z;
#pragma unroll
    for (int r = 0; r < 8; ++r) m_r[r] = -__builtin_inff();
  }

  const _Float16* Kg = Kh + (size_t)h * LSEQ * DK;           // [key][d]
  const _Float16* Vg = VhT + (size_t)h * (size_t)DK * LSEQ;  // [d][key]
  _Float16* Pw = Ps + wave * (16 * 64);

  int buf = 0;
  stage_issue(Kg, Vg, Ks[0], Vs[0], tid, wave);

  for (int kv = 0; kv < LSEQ; kv += 64) {
    stage_sync(wave);  // TENSORcnt/ASYNCcnt drained by owner, then workgroup barrier
    if (kv + 64 < LSEQ)
      stage_issue(Kg + (size_t)(kv + 64) * DK, Vg + (kv + 64),
                  Ks[buf ^ 1], Vs[buf ^ 1], tid, wave);

    const _Float16* Ksb = Ks[buf];
    const _Float16* Vsb = Vs[buf];

    // S = (Q*scale) K^T : 16 x 64 in four 16x16 f32 tiles
    v8f s[4];
#pragma unroll
    for (int nt = 0; nt < 4; ++nt) {
      const _Float16* kp = Ksb + (size_t)(nt * 16 + ml) * 64 + (hl << 4);
      v16h b0 = *(const v16h*)(kp);
      v16h b1 = *(const v16h*)(kp + 32);
      v8f c = {};
      c = wmma_f16(aq0, b0, c);
      c = wmma_f16(aq1, b1, c);
      s[nt] = c;
    }

    // online softmax (max only; sums ride the ones-column WMMA)
#pragma unroll
    for (int r = 0; r < 8; ++r) {
      float v = fmaxf(fmaxf(s[0][r], s[1][r]), fmaxf(s[2][r], s[3][r]));
      v = rowmax16(v);
      const float mn = fmaxf(m_r[r], v);
      const float corr = __expf(m_r[r] - mn);
      m_r[r] = mn;
#pragma unroll
      for (int nt = 0; nt < 4; ++nt) s[nt][r] = __expf(s[nt][r] - mn);
#pragma unroll
      for (int d = 0; d < 5; ++d) o[d][r] *= corr;
    }

    // restage P (f16, row-major 16x64) for A-fragment reload
#pragma unroll
    for (int nt = 0; nt < 4; ++nt)
#pragma unroll
      for (int r = 0; r < 8; ++r)
        Pw[(size_t)(r + (hl << 3)) * 64 + nt * 16 + ml] = (_Float16)s[nt][r];
    __syncthreads();

    // O += P V ; denominator += P * ones
    const _Float16* pp = Pw + (size_t)ml * 64 + (hl << 4);
    v16h ap0 = load_a_frag(pp);
    v16h ap1 = load_a_frag(pp + 32);
#pragma unroll
    for (int d = 0; d < 4; ++d) {
      const _Float16* vp = Vsb + (size_t)(d * 16 + ml) * 64 + (hl << 4);
      v16h b0 = *(const v16h*)(vp);
      v16h b1 = *(const v16h*)(vp + 32);
      o[d] = wmma_f16(ap0, b0, o[d]);
      o[d] = wmma_f16(ap1, b1, o[d]);
    }
    o[4] = wmma_f16(ap0, bone, o[4]);
    o[4] = wmma_f16(ap1, bone, o[4]);

    buf ^= 1;
  }

  // epilogue: O /= rowsum (broadcast from column 0 of the denominator tile)
#pragma unroll
  for (int r = 0; r < 8; ++r) {
    const float l = __shfl(o[4][r], lane & 16);  // ml==0 lane of this half-group
    const float inv = 1.0f / l;
#pragma unroll
    for (int d = 0; d < 4; ++d) {
      const float v = o[d][r] * inv;
      Oh[(size_t)(q0 + r + (hl << 3)) * DMODEL + h * DK + d * 16 + ml] = (_Float16)v;
    }
  }
}

// ---- host side ----

extern "C" void kernel_launch(void* const* d_in, const int* in_sizes, int n_in,
                              void* d_out, int out_size, void* d_ws, size_t ws_size,
                              hipStream_t stream) {
  (void)in_sizes; (void)n_in; (void)out_size; (void)ws_size;
  const float* query = (const float*)d_in[0];
  const float* key   = (const float*)d_in[1];
  const float* value = (const float*)d_in[2];
  const float* Wq = (const float*)d_in[3];
  const float* Wk = (const float*)d_in[4];
  const float* Wv = (const float*)d_in[5];
  const float* Wo = (const float*)d_in[6];
  const float* bq = (const float*)d_in[7];
  const float* bk = (const float*)d_in[8];
  const float* bv = (const float*)d_in[9];
  const float* bo = (const float*)d_in[10];

  const size_t LD = (size_t)LSEQ * DMODEL;
  const size_t MM = (size_t)DMODEL * DMODEL;

  _Float16* w  = (_Float16*)d_ws;
  _Float16* xq = w;            _Float16* xk = xq + LD;  _Float16* xv = xk + LD;
  _Float16* wq = xv + LD;      _Float16* wk = wq + MM;
  _Float16* wv = wk + MM;      _Float16* wo = wv + MM;
  _Float16* Qh = wo + MM;      _Float16* Kh = Qh + LD;
  _Float16* Vh = Kh + LD;      _Float16* Oh = Vh + LD;   // 64 MiB f16 total

  cvt_f32_f16<<<(int)(LD / 256), 256, 0, stream>>>(query, xq, (int)LD);
  cvt_f32_f16<<<(int)(LD / 256), 256, 0, stream>>>(key,   xk, (int)LD);
  cvt_f32_f16<<<(int)(LD / 256), 256, 0, stream>>>(value, xv, (int)LD);
  cvt_f32_f16<<<(int)(MM / 256), 256, 0, stream>>>(Wq, wq, (int)MM);
  cvt_f32_f16<<<(int)(MM / 256), 256, 0, stream>>>(Wk, wk, (int)MM);
  cvt_f32_f16<<<(int)(MM / 256), 256, 0, stream>>>(Wv, wv, (int)MM);
  cvt_f32_f16<<<(int)(MM / 256), 256, 0, stream>>>(Wo, wo, (int)MM);

  const int gblocks = (LSEQ / 16) * (DMODEL / 16) / 8;  // 2048
  gemm16_wmma<0><<<gblocks, 256, 0, stream>>>(xq, wq, bq, Qh);
  gemm16_wmma<0><<<gblocks, 256, 0, stream>>>(xk, wk, bk, Kh);
  gemm16_wmma<1><<<gblocks, 256, 0, stream>>>(xv, wv, bv, Vh);  // V stored [H][DK][L]

  attn_wmma<<<dim3(LSEQ / 64, NHEAD), 128, 0, stream>>>(Qh, Kh, Vh, Oh);

  gemm16_wmma<2><<<gblocks, 256, 0, stream>>>(Oh, wo, bo, d_out);
}
